// GeometryKernelAttention_481036337896
// MI455X (gfx1250) — matrix operand order repaired and checked by reading, not compiled, not run on hardware
//
#include <hip/hip_runtime.h>
#include <math.h>

// ---------------- problem constants ----------------
#define CB   4      // batch
#define CN   6      // cameras
#define CQ   2500   // BEV queries (50*50)
#define CG   9      // kernel samples (3x3)
#define CD   128    // DIM == FEAT_DIM == HEADS*DIM_HEAD
#define CFH  14
#define CFW  30
#define CP   (CFH*CFW)      // 420 pixels
#define CQT  157            // ceil(2500/16)
#define ATT_SCALE 0.17677669529663687f   // 32^-0.5

typedef __attribute__((ext_vector_type(16))) __bf16 bf16x16;
typedef __attribute__((ext_vector_type(8)))  float  floatx8;

// ---------------- WMMA helpers (gfx1250, wave32) ----------------
__device__ __forceinline__ floatx8 wmma_bf16(bf16x16 a, bf16x16 b, floatx8 c) {
  return __builtin_amdgcn_wmma_f32_16x16x32_bf16(false, a, false, b, (short)0, c,
                                                 false, false);
}

// 16-row GEMM tile: sOut[16 x cols] (+= handled by caller) = A(16xK,bf16 LDS) @ Wp(packed bf16)
// Waves split the N-tiles [nt0, nt1); all lanes active -> EXEC all ones for WMMA.
__device__ __forceinline__ void gemm16(const __bf16* sA, int lda,
                                       const __bf16* Wp, int K, int NT,
                                       int nt0, int nt1,
                                       float* sOut, int ldc, int colBase,
                                       int lane, int wv) {
  const int KC = K >> 5;
  const int m   = lane & 15;
  const int off = (lane >> 4) * 8;
  for (int nt = nt0 + wv; nt < nt1; nt += 4) {
    floatx8 acc;
#pragma unroll
    for (int r = 0; r < 8; ++r) acc[r] = 0.f;
    for (int kc = 0; kc < KC; ++kc) {
      // A fragment: 16-bit A 16x32 layout (ISA 7.12.2): lane m=lane&15,
      // halves 0..7 -> K = kc*32+off+i, halves 8..15 -> K = kc*32+16+off+i
      const __bf16* ap = sA + m * lda + kc * 32 + off;
      bf16x16 a;
#pragma unroll
      for (int i = 0; i < 8; ++i) { a[i] = ap[i]; a[8 + i] = ap[16 + i]; }
      // B fragment: pre-packed so each lane reads 16 contiguous bf16 (32B)
      const __bf16* bp = Wp + ((((size_t)kc * NT + nt) * 32 + lane) << 4);
      bf16x16 b;
#pragma unroll
      for (int i = 0; i < 16; ++i) b[i] = bp[i];
      acc = wmma_bf16(a, b, acc);
    }
    // C/D f32 16x16 layout: vgpr r -> m = r + (lane>>4)*8, n = lane&15
    const int n  = (lane & 15) + (nt << 4) - colBase;
    const int mb = (lane >> 4) * 8;
#pragma unroll
    for (int r = 0; r < 8; ++r) sOut[(mb + r) * ldc + n] = acc[r];
  }
}

// Row-wise LayerNorm over 16 rows x 128 cols, 128 threads (8 per row).
// Optionally writes bf16 (for WMMA A) and/or f32 (may alias the input).
__device__ __forceinline__ void ln16(const float* sInp, int ldi,
                                     const float* w, const float* bias,
                                     __bf16* bOut, int ldb,
                                     float* fOut, int ldf,
                                     int doRelu, float* sRed, int tid) {
  const int r = tid >> 3, c = tid & 7;
  float s = 0.f, s2 = 0.f;
  for (int j = c; j < 128; j += 8) { float v = sInp[r * ldi + j]; s += v; s2 += v * v; }
  sRed[tid] = s; sRed[128 + tid] = s2;
  __syncthreads();
  if (c == 0) {
    float ts = 0.f, ts2 = 0.f;
    for (int j = 0; j < 8; ++j) { ts += sRed[r * 8 + j]; ts2 += sRed[128 + r * 8 + j]; }
    float mu  = ts * 0.0078125f;
    float var = ts2 * 0.0078125f - mu * mu;
    sRed[256 + r] = mu;
    sRed[272 + r] = rsqrtf(var + 1e-5f);
  }
  __syncthreads();
  const float mu = sRed[256 + r], inv = sRed[272 + r];
  for (int j = c; j < 128; j += 8) {
    float y = (sInp[r * ldi + j] - mu) * inv * w[j] + bias[j];
    if (doRelu) y = fmaxf(y, 0.f);
    if (bOut) bOut[r * ldb + j] = (__bf16)y;
    if (fOut) fOut[r * ldf + j] = y;
  }
  __syncthreads();
}

// ---------------- prep kernels ----------------

// Pack f32 (K x N) weight into per-lane-contiguous bf16 B fragments.
// element i of lane -> n = (lane&15)+nt*16, k = kc*32 + (i>>3)*16 + (lane>>4)*8 + (i&7)
__global__ void gka_pack(const float* __restrict__ src, __bf16* __restrict__ dst,
                         int K, int N) {
  int i = blockIdx.x * 256 + threadIdx.x;
  if (i >= K * N) return;
  int i4   = i & 15;
  int lane = (i >> 4) & 31;
  int t    = i >> 9;
  int NT   = N >> 4;
  int nt = t % NT, kc = t / NT;
  int n = (lane & 15) + (nt << 4);
  int k = (kc << 5) + ((i4 >> 3) << 4) + ((lane >> 4) << 3) + (i4 & 7);
  dst[i] = (__bf16)src[k * N + n];
}

// c_embed (b,n,128) and P4 = I4 @ E  (per camera)
__global__ void gka_prep(const float* __restrict__ cam_w, const float* __restrict__ Imat,
                         const float* __restrict__ Emat, const float* __restrict__ Einv,
                         float* __restrict__ c_embed, float* __restrict__ P4) {
  int bn = blockIdx.x, tid = threadIdx.x;
  const float* Ei = Einv + bn * 16;
  {
    float c0 = Ei[3], c1 = Ei[7], c2 = Ei[11], c3 = Ei[15];
    const float* cw = cam_w + tid * 4;
    c_embed[bn * 128 + tid] = cw[0] * c0 + cw[1] * c1 + cw[2] * c2 + cw[3] * c3;
  }
  if (tid < 16) {
    int i = tid >> 2, j = tid & 3;
    const float* Ip = Imat + bn * 9;
    const float* Ep = Emat + bn * 16;
    float s = 0.f;
    for (int k = 0; k < 4; ++k) {
      float a = (i < 3 && k < 3) ? Ip[i * 3 + k] : ((i == 3 && k == 3) ? 1.f : 0.f);
      s += a * Ep[k * 4 + j];
    }
    P4[bn * 16 + i * 4 + j] = s;
  }
}

// w_embed[q][d] = bev_w @ bev_grid[:2] + bev_b
__global__ void gka_wembed(const float* __restrict__ bev_grid, const float* __restrict__ bev_w,
                           const float* __restrict__ bev_b, float* __restrict__ wembed) {
  int idx = blockIdx.x * 256 + threadIdx.x;
  if (idx >= CQ * CD) return;
  int q = idx >> 7, d = idx & 127;
  float gx = bev_grid[q], gy = bev_grid[CQ + q];
  wembed[idx] = bev_w[d * 2] * gx + bev_w[d * 2 + 1] * gy + bev_b[d];
}

// img_embed (b,n,p,128) normalized over d; stored row-major per pixel for gathers
__global__ void gka_imgembed(const float* __restrict__ Iinv, const float* __restrict__ Einv,
                             const float* __restrict__ img_w, const float* __restrict__ c_embed,
                             float* __restrict__ imge) {
  int bid = blockIdx.x, tid = threadIdx.x;
  int bn = bid / CP, pix = bid % CP;
  int ph = pix / CFW, pw = pix % CFW;
  float px = (float)pw * (480.f / (CFW - 1));
  float py = (float)ph * (224.f / (CFH - 1));
  const float* Ii = Iinv + bn * 9;
  const float* Ei = Einv + bn * 16;
  float cam0 = Ii[0] * px + Ii[1] * py + Ii[2];
  float cam1 = Ii[3] * px + Ii[4] * py + Ii[5];
  float cam2 = Ii[6] * px + Ii[7] * py + Ii[8];
  float d0 = Ei[0] * cam0 + Ei[1] * cam1 + Ei[2] * cam2 + Ei[3];
  float d1 = Ei[4] * cam0 + Ei[5] * cam1 + Ei[6] * cam2 + Ei[7];
  float d2 = Ei[8] * cam0 + Ei[9] * cam1 + Ei[10] * cam2 + Ei[11];
  float d3 = Ei[12] * cam0 + Ei[13] * cam1 + Ei[14] * cam2 + Ei[15];
  const float* iw = img_w + tid * 4;
  float e = iw[0] * d0 + iw[1] * d1 + iw[2] * d2 + iw[3] * d3 - c_embed[bn * 128 + tid];
  __shared__ float red[128];
  red[tid] = e * e;
  __syncthreads();
  for (int s = 64; s > 0; s >>= 1) { if (tid < s) red[tid] += red[tid + s]; __syncthreads(); }
  float inv = 1.f / (sqrtf(red[0]) + 1e-7f);
  imge[((size_t)bn * CP + pix) * 128 + tid] = e * inv;
}

// 1x7 conv (pad 3 in width), output (b,n,p,128)
__global__ void gka_conv(const float* __restrict__ feature, const float* __restrict__ kw,
                         const float* __restrict__ kb, float* __restrict__ featc) {
  int bid = blockIdx.x, tid = threadIdx.x;
  int bn = bid / CP, pix = bid % CP;
  int h = pix / CFW, w0 = pix % CFW;
  __shared__ float win[128 * 7];
  for (int t = 0; t < 7; ++t) {
    int ww = w0 + t - 3;
    win[tid * 7 + t] = (ww >= 0 && ww < CFW)
        ? feature[((size_t)(bn * 128 + tid) * CFH + h) * CFW + ww] : 0.f;
  }
  __syncthreads();
  float acc = kb[tid];
  for (int ic = 0; ic < 128; ++ic) {
    const float* wr = kw + ((size_t)tid * 128 + ic) * 7;
    const float* xr = win + ic * 7;
#pragma unroll
    for (int t = 0; t < 7; ++t) acc += xr[t] * wr[t];
  }
  featc[((size_t)bn * CP + pix) * 128 + tid] = acc;
}

// project BEV points -> per (b,n,q): mask + 9 gather indices
__global__ void gka_project(const float* __restrict__ bev_grid, const float* __restrict__ P4,
                            int* __restrict__ inds, int* __restrict__ maskb) {
  int idx = blockIdx.x * 256 + threadIdx.x;
  if (idx >= CB * CN * CQ) return;
  int bn = idx / CQ, q = idx % CQ;
  float p0 = bev_grid[q], p1 = bev_grid[CQ + q];
  const float* M = P4 + bn * 16;
  float sp0 = M[0] * p0 + M[1] * p1 - M[2] + M[3];
  float sp1 = M[4] * p0 + M[5] * p1 - M[6] + M[7];
  float sp2 = M[8] * p0 + M[9] * p1 - M[10] + M[11];
  float zm = fmaxf(sp2, 1e-5f);
  float u = sp0 / zm * (1.f / 480.f);
  float v = sp1 / zm * (1.f / 224.f);
  int m = (sp2 > 1e-5f) && (u > 0.f) && (u < 1.f) && (v > 0.f) && (v < 1.f);
  maskb[idx] = m;
  float uc = fminf(fmaxf(u, -1.f), 2.f), vc = fminf(fmaxf(v, -1.f), 2.f);
  int iu0 = (int)rintf(uc * (float)CFW);
  int iv0 = (int)rintf(vc * (float)CFH);
  for (int g = 0; g < 9; ++g) {
    int iu = iu0 + (g % 3) - 1, iv = iv0 + (g / 3) - 1;
    iu = min(max(iu, 0), CFW - 1);
    iv = min(max(iv, 0), CFH - 1);
    inds[(size_t)idx * CG + g] = iv * CFW + iu;
  }
}

// ---------------- fused attention mega-kernel ----------------
// One block owns 16 queries of one batch. Per camera: q-proj; per g-tile:
// gather -> LN(+relu) -> WMMA(fp/fl) -> LN -> WMMA(k/v) -> online softmax.
// Then proj + pre-LN + MLP(gelu) + post-LN, all on the same 16-row tile.
__global__ void __launch_bounds__(128)
gka_attn(const float* __restrict__ x, const float* __restrict__ c_embed,
         const float* __restrict__ wembed, const float* __restrict__ imge,
         const float* __restrict__ featc, const int* __restrict__ inds,
         const int* __restrict__ maskb,
         const __bf16* __restrict__ pw_fp, const __bf16* __restrict__ pw_fl,
         const __bf16* __restrict__ pw_k,  const __bf16* __restrict__ pw_v,
         const __bf16* __restrict__ pw_q,  const __bf16* __restrict__ pw_proj,
         const __bf16* __restrict__ pw_m1, const __bf16* __restrict__ pw_m2,
         const float* __restrict__ fp_ln_w, const float* __restrict__ fp_ln_b,
         const float* __restrict__ fl_ln_w, const float* __restrict__ fl_ln_b,
         const float* __restrict__ q_ln_w,  const float* __restrict__ q_ln_b,
         const float* __restrict__ q_b,
         const float* __restrict__ k_ln_w,  const float* __restrict__ k_ln_b,
         const float* __restrict__ k_b,
         const float* __restrict__ v_ln_w,  const float* __restrict__ v_ln_b,
         const float* __restrict__ v_b,
         const float* __restrict__ proj_b,
         const float* __restrict__ pre_ln_w, const float* __restrict__ pre_ln_b,
         const float* __restrict__ mlp_b1,   const float* __restrict__ mlp_b2,
         const float* __restrict__ post_ln_w, const float* __restrict__ post_ln_b,
         float* __restrict__ out) {
  const int blk = blockIdx.x;
  const int b = blk / CQT, qt = blk % CQT;
  const int q0 = qt * 16;
  const int tid = threadIdx.x;
  const int lane = tid & 31, wv = tid >> 5;

  __shared__ float  sWe[16 * 128];     // w_embed rows
  __shared__ float  sX [16 * 128];     // x rows, later z residual
  __shared__ float  sIn[16 * 132];     // f32 staging / gemm out
  __shared__ float  sQ [16 * 128];     // projected q rows (per cam)
  __shared__ float  sBig[16 * 128];    // img rows; later aliased as 16x256 bf16 H
  __shared__ __bf16 sA1[16 * 128];
  __shared__ __bf16 sA2[16 * 128];
  __shared__ __bf16 sKt[16 * 128];
  __shared__ __bf16 sVt[16 * 128];
  __shared__ float  sRed[304];
  __shared__ float  sCe[128];

  // preload w_embed and x rows for this query tile (clamp tail)
  for (int i = tid; i < 2048; i += 128) {
    int r = i >> 7, d = i & 127;
    int q = q0 + r; if (q >= CQ) q = CQ - 1;
    sWe[r * 128 + d] = wembed[q * 128 + d];
    sX [r * 128 + d] = x[((size_t)(b * 128 + d)) * CQ + q];
  }

  // online softmax state (threads 0..63: one (query,head) each)
  float accv[32];
#pragma unroll
  for (int d = 0; d < 32; ++d) accv[d] = 0.f;
  float mrun = -1e30f, lrun = 0.f;

  for (int n = 0; n < CN; ++n) {
    const int bn = b * CN + n;
    sCe[tid] = c_embed[bn * 128 + tid];
    __syncthreads();

    // ---- query rows: normalize(w_embed - c_embed) + x  -> sIn ----
    {
      int r = tid >> 3, c = tid & 7;
      float s2 = 0.f;
      for (int j = c; j < 128; j += 8) { float t = sWe[r * 128 + j] - sCe[j]; s2 += t * t; }
      sRed[tid] = s2;
      __syncthreads();
      if (c == 0) {
        float ts = 0.f;
        for (int j = 0; j < 8; ++j) ts += sRed[r * 8 + j];
        sRed[256 + r] = 1.f / (sqrtf(ts) + 1e-7f);
      }
      __syncthreads();
      float inv = sRed[256 + r];
      for (int j = c; j < 128; j += 8)
        sIn[r * 132 + j] = (sWe[r * 128 + j] - sCe[j]) * inv + sX[r * 128 + j];
      __syncthreads();
    }
    // q = LN(query) @ q_w + q_b
    ln16(sIn, 132, q_ln_w, q_ln_b, sA1, 128, nullptr, 0, 0, sRed, tid);
    gemm16(sA1, 128, pw_q, 128, 8, 0, 8, sQ, 128, 0, lane, wv);
    __syncthreads();
    for (int i = tid; i < 2048; i += 128) sQ[i] += q_b[i & 127];
    __syncthreads();

    for (int g = 0; g < CG; ++g) {
      // ---- gather sampled feat/img rows ----
      for (int i = tid; i < 2048; i += 128) {
        int r = i >> 7, d = i & 127;
        int q = q0 + r; if (q >= CQ) q = CQ - 1;
        int p = inds[((size_t)bn * CQ + q) * CG + g];
        sIn [r * 132 + d] = featc[((size_t)bn * CP + p) * 128 + d];
        sBig[r * 128 + d] = imge [((size_t)bn * CP + p) * 128 + d];
      }
      __syncthreads();
      ln16(sIn, 132, fp_ln_w, fp_ln_b, sA1, 128, nullptr, 0, 1, sRed, tid);
      ln16(sIn, 132, fl_ln_w, fl_ln_b, sA2, 128, nullptr, 0, 1, sRed, tid);

      // value path: v = LN_v(A_fl @ fl_w) @ v_w + v_b
      gemm16(sA2, 128, pw_fl, 128, 8, 0, 8, sIn, 132, 0, lane, wv);
      __syncthreads();
      ln16(sIn, 132, v_ln_w, v_ln_b, sA2, 128, nullptr, 0, 0, sRed, tid);
      gemm16(sA2, 128, pw_v, 128, 8, 0, 8, sIn, 132, 0, lane, wv);
      __syncthreads();
      for (int i = tid; i < 2048; i += 128) {
        int r = i >> 7, d = i & 127;
        sVt[r * 128 + d] = (__bf16)(sIn[r * 132 + d] + v_b[d]);
      }
      __syncthreads();

      // key path: k = LN_k(img + A_fp @ fp_w) @ k_w + k_b
      gemm16(sA1, 128, pw_fp, 128, 8, 0, 8, sIn, 132, 0, lane, wv);
      __syncthreads();
      for (int i = tid; i < 2048; i += 128) {
        int r = i >> 7, d = i & 127;
        sIn[r * 132 + d] += sBig[r * 128 + d];
      }
      __syncthreads();
      ln16(sIn, 132, k_ln_w, k_ln_b, sA1, 128, nullptr, 0, 0, sRed, tid);
      gemm16(sA1, 128, pw_k, 128, 8, 0, 8, sIn, 132, 0, lane, wv);
      __syncthreads();
      for (int i = tid; i < 2048; i += 128) {
        int r = i >> 7, d = i & 127;
        sKt[r * 128 + d] = (__bf16)(sIn[r * 132 + d] + k_b[d]);
      }
      __syncthreads();

      // ---- online softmax update for this (n,g) ----
      if (tid < 64) {
        int qi = tid >> 2, h = tid & 3;
        int q = q0 + qi; if (q >= CQ) q = CQ - 1;
        int mk = maskb[(size_t)bn * CQ + q];
        const float* qp = &sQ[qi * 128 + h * 32];
        float s = 0.f;
#pragma unroll
        for (int d = 0; d < 32; ++d) s += qp[d] * (float)sKt[qi * 128 + h * 32 + d];
        s = mk ? s * ATT_SCALE : -1e9f;
        float mnew = fmaxf(mrun, s);
        float corr = __expf(mrun - mnew);
        float p    = __expf(s - mnew);
        lrun = lrun * corr + p;
#pragma unroll
        for (int d = 0; d < 32; ++d)
          accv[d] = accv[d] * corr + p * (float)sVt[qi * 128 + h * 32 + d];
        mrun = mnew;
      }
      __syncthreads();
    }
  }

  // ---- attention output rows ----
  if (tid < 64) {
    int qi = tid >> 2, h = tid & 3;
    float invl = 1.f / lrun;
#pragma unroll
    for (int d = 0; d < 32; ++d) sIn[qi * 132 + h * 32 + d] = accv[d] * invl;
  }
  __syncthreads();
  for (int i = tid; i < 2048; i += 128) {
    int r = i >> 7, d = i & 127;
    sA1[r * 128 + d] = (__bf16)sIn[r * 132 + d];
  }
  __syncthreads();

  // z1 = a @ proj_w + proj_b + x
  gemm16(sA1, 128, pw_proj, 128, 8, 0, 8, sIn, 132, 0, lane, wv);
  __syncthreads();
  for (int i = tid; i < 2048; i += 128) {
    int r = i >> 7, d = i & 127;
    sX[r * 128 + d] = sIn[r * 132 + d] + proj_b[d] + sX[r * 128 + d];
  }
  __syncthreads();
  // z2 = LN_pre(z1)  (f32 in-place into sX, bf16 into sA1)
  ln16(sX, 128, pre_ln_w, pre_ln_b, sA1, 128, sX, 128, 0, sRed, tid);

  // MLP: h = gelu(z2 @ w1 + b1) ; z3 = z2 + h @ w2 + b2
  __bf16* sH = (__bf16*)sBig;  // 16 x 256 bf16 aliases img staging
  for (int half = 0; half < 2; ++half) {
    gemm16(sA1, 128, pw_m1, 128, 16, half * 8, half * 8 + 8, sIn, 132, half * 128, lane, wv);
    __syncthreads();
    for (int i = tid; i < 2048; i += 128) {
      int r = i >> 7, d = i & 127;
      float v = sIn[r * 132 + d] + mlp_b1[half * 128 + d];
      float gl = 0.5f * v * (1.f + erff(v * 0.70710678118654752f));
      sH[r * 256 + half * 128 + d] = (__bf16)gl;
    }
    __syncthreads();
  }
  gemm16(sH, 256, pw_m2, 256, 8, 0, 8, sIn, 132, 0, lane, wv);
  __syncthreads();
  for (int i = tid; i < 2048; i += 128) {
    int r = i >> 7, d = i & 127;
    sX[r * 128 + d] = sX[r * 128 + d] + sIn[r * 132 + d] + mlp_b2[d];
  }
  __syncthreads();
  // out = LN_post(z3), stored as (b, DIM, H, W)
  ln16(sX, 128, post_ln_w, post_ln_b, nullptr, 0, sX, 128, 0, sRed, tid);
  for (int i = tid; i < 2048; i += 128) {
    int r = i >> 7, d = i & 127;
    int q = q0 + r;
    if (q < CQ) out[((size_t)(b * 128 + d)) * CQ + q] = sX[r * 128 + d];
  }
}

// ---------------- host launcher ----------------
extern "C" void kernel_launch(void* const* d_in, const int* in_sizes, int n_in,
                              void* d_out, int out_size, void* d_ws, size_t ws_size,
                              hipStream_t stream) {
  const float* x        = (const float*)d_in[0];
  const float* bev_grid = (const float*)d_in[1];
  const float* feature  = (const float*)d_in[2];
  const float* Imat     = (const float*)d_in[3];
  const float* Emat     = (const float*)d_in[4];
  const float* I_inv    = (const float*)d_in[5];
  const float* E_inv    = (const float*)d_in[6];
  const float* cam_w    = (const float*)d_in[7];
  const float* img_w    = (const float*)d_in[8];
  const float* bev_w    = (const float*)d_in[9];
  const float* bev_b    = (const float*)d_in[10];
  const float* kconv_w  = (const float*)d_in[11];
  const float* kconv_b  = (const float*)d_in[12];
  const float* fp_ln_w  = (const float*)d_in[13];
  const float* fp_ln_b  = (const float*)d_in[14];
  const float* fp_w     = (const float*)d_in[15];
  const float* fl_ln_w  = (const float*)d_in[16];
  const float* fl_ln_b  = (const float*)d_in[17];
  const float* fl_w     = (const float*)d_in[18];
  const float* q_ln_w   = (const float*)d_in[19];
  const float* q_ln_b   = (const float*)d_in[20];
  const float* q_w      = (const float*)d_in[21];
  const float* q_b      = (const float*)d_in[22];
  const float* k_ln_w   = (const float*)d_in[23];
  const float* k_ln_b   = (const float*)d_in[24];
  const float* k_w      = (const float*)d_in[25];
  const float* k_b      = (const float*)d_in[26];
  const float* v_ln_w   = (const float*)d_in[27];
  const float* v_ln_b   = (const float*)d_in[28];
  const float* v_w      = (const float*)d_in[29];
  const float* v_b      = (const float*)d_in[30];
  const float* proj_w   = (const float*)d_in[31];
  const float* proj_b   = (const float*)d_in[32];
  const float* pre_ln_w = (const float*)d_in[33];
  const float* pre_ln_b = (const float*)d_in[34];
  const float* mlp_w1   = (const float*)d_in[35];
  const float* mlp_b1   = (const float*)d_in[36];
  const float* mlp_w2   = (const float*)d_in[37];
  const float* mlp_b2   = (const float*)d_in[38];
  const float* post_ln_w= (const float*)d_in[39];
  const float* post_ln_b= (const float*)d_in[40];
  float* out = (float*)d_out;

  char* ws = (char*)d_ws;
  size_t o = 0;
  auto take = [&](size_t bytes) -> char* {
    char* p = ws + o;
    o += (bytes + 255) & ~(size_t)255;
    return p;
  };
  __bf16* pw_fp   = (__bf16*)take(128 * 128 * 2);
  __bf16* pw_fl   = (__bf16*)take(128 * 128 * 2);
  __bf16* pw_k    = (__bf16*)take(128 * 128 * 2);
  __bf16* pw_v    = (__bf16*)take(128 * 128 * 2);
  __bf16* pw_q    = (__bf16*)take(128 * 128 * 2);
  __bf16* pw_proj = (__bf16*)take(128 * 128 * 2);
  __bf16* pw_m1   = (__bf16*)take(128 * 256 * 2);
  __bf16* pw_m2   = (__bf16*)take(256 * 128 * 2);
  float* c_embed = (float*)take(CB * CN * 128 * 4);
  float* P4      = (float*)take(CB * CN * 16 * 4);
  float* wembed  = (float*)take((size_t)CQ * 128 * 4);
  float* imge    = (float*)take((size_t)CB * CN * CP * 128 * 4);
  float* featc   = (float*)take((size_t)CB * CN * CP * 128 * 4);
  int*   inds    = (int*)take((size_t)CB * CN * CQ * CG * 4);
  int*   maskb   = (int*)take((size_t)CB * CN * CQ * 4);

  // pack all GEMM weights into bf16 B-fragment layout
  gka_pack<<<(128 * 128 + 255) / 256, 256, 0, stream>>>(fp_w,   pw_fp,   128, 128);
  gka_pack<<<(128 * 128 + 255) / 256, 256, 0, stream>>>(fl_w,   pw_fl,   128, 128);
  gka_pack<<<(128 * 128 + 255) / 256, 256, 0, stream>>>(k_w,    pw_k,    128, 128);
  gka_pack<<<(128 * 128 + 255) / 256, 256, 0, stream>>>(v_w,    pw_v,    128, 128);
  gka_pack<<<(128 * 128 + 255) / 256, 256, 0, stream>>>(q_w,    pw_q,    128, 128);
  gka_pack<<<(128 * 128 + 255) / 256, 256, 0, stream>>>(proj_w, pw_proj, 128, 128);
  gka_pack<<<(128 * 256 + 255) / 256, 256, 0, stream>>>(mlp_w1, pw_m1,   128, 256);
  gka_pack<<<(256 * 128 + 255) / 256, 256, 0, stream>>>(mlp_w2, pw_m2,   256, 128);

  gka_prep<<<CB * CN, 128, 0, stream>>>(cam_w, Imat, Emat, E_inv, c_embed, P4);
  gka_wembed<<<(CQ * 128 + 255) / 256, 256, 0, stream>>>(bev_grid, bev_w, bev_b, wembed);
  gka_imgembed<<<CB * CN * CP, 128, 0, stream>>>(I_inv, E_inv, img_w, c_embed, imge);
  gka_conv<<<CB * CN * CP, 128, 0, stream>>>(feature, kconv_w, kconv_b, featc);
  gka_project<<<(CB * CN * CQ + 255) / 256, 256, 0, stream>>>(bev_grid, P4, inds, maskb);

  gka_attn<<<CB * CQT, 128, 0, stream>>>(
      x, c_embed, wembed, imge, featc, inds, maskb,
      pw_fp, pw_fl, pw_k, pw_v, pw_q, pw_proj, pw_m1, pw_m2,
      fp_ln_w, fp_ln_b, fl_ln_w, fl_ln_b,
      q_ln_w, q_ln_b, q_b, k_ln_w, k_ln_b, k_b, v_ln_w, v_ln_b, v_b,
      proj_b, pre_ln_w, pre_ln_b, mlp_b1, mlp_b2, post_ln_w, post_ln_b, out);
}
